// GINERegressionModel_3143916060997
// MI455X (gfx1250) — compile-verified, matrix operand order checked by reading, not run on hardware
//
#include <hip/hip_runtime.h>

// ---------------- problem constants ----------------
#define N_NODES 100000
#define N_EDGES 1600000
#define HID     64
#define LAYERS  4
#define NODE_D  32
#define EDGE_D  16
#define GF_D    32
#define NGRAPH  1024
#define JKDIM   (HID * (LAYERS + 1))   // 320
#define TOTDIM  (JKDIM + GF_D)         // 352
#define BN_EPS  1e-5f
#define GN_EPS  1e-5f

typedef __bf16 bf16_t;
typedef __attribute__((ext_vector_type(16))) __bf16 v16bf;
typedef __attribute__((ext_vector_type(8)))  __bf16 v8bf;
typedef __attribute__((ext_vector_type(4)))  __bf16 v4bf;
typedef __attribute__((ext_vector_type(8)))  float  v8f;

// Native CDNA5 fp32 global atomic add (no CAS loop).
__device__ __forceinline__ void atomic_add_f32(float* p, float v) {
  asm volatile("global_atomic_add_f32 %0, %1, off" :: "v"(p), "v"(v) : "memory");
}

// ---------------------------------------------------------------------------
// WMMA GEMM, compile-time K and NT(=outN/16):
//   C[M, NT*16] = act(A[M,K] @ W[K,NT*16] + bias)
// 128-thread block = 4 waves; wave w owns row block (blockIdx.x*4 + w) and
// computes ALL NT column tiles from one A fragment (A read exactly once).
// W is staged per-block into LDS as bf16, pre-swizzled into WMMA B-fragment
// order: per (k-step, tile) two 16B half-regions with 16B lane stride, so
// each lane's fragment is two ds_load_b128 at the 2-pass LDS bank minimum.
// K values not multiple of 32 (K=16) are zero-padded at staging time (B) and
// by the compile-time run guard (A).
// ---------------------------------------------------------------------------
template<int K, int NT, int ACT, int OUTBF16>
__global__ void k_gemm(const float* __restrict__ A, int lda,
                       const float* __restrict__ W, int wld,
                       const float* __restrict__ bias,
                       void* __restrict__ Cout, int ldc, int M)
{
  constexpr int KT = (K + 31) / 32;
  // per (t,nt): 2 half-regions * 32 lanes * 8 bf16 = 512 elements (1KB)
  __shared__ __bf16 smem[KT * NT * 512];

  // ---- cooperative stage: W[k][col] -> fragment-order bf16 in LDS ----
  for (int idx = threadIdx.x; idx < KT * 32 * NT * 16; idx += blockDim.x) {
    const int k   = idx / (NT * 16);
    const int col = idx % (NT * 16);
    const float wv = (k < K) ? W[k * wld + col] : 0.0f;
    const int t    = k >> 5;
    const int nt   = col >> 4;
    const int lane = (col & 15) + (((k & 15) >> 3) << 4);
    const int e    = (((k & 31) >> 4) << 3) + (k & 7);
    smem[((((t * NT + nt) * 2 + (e >> 3)) * 32 + lane) << 3) + (e & 7)] = (__bf16)wv;
  }
  __syncthreads();

  const int lane = threadIdx.x & 31;
  const int wave = threadIdx.x >> 5;
  const int rb   = blockIdx.x * 4 + wave;       // row block (16 rows)
  if (rb * 16 >= M) return;                     // wave-uniform: EXEC stays all-1s

  const int mrow = lane & 15;
  const int hk   = (lane >> 4) << 3;            // lane-half offset (K for A/B, M for C/D)
  const float* __restrict__ ap = A + ((long long)rb * 16 + mrow) * lda;
  const __bf16* __restrict__ bp = smem + lane * 8;

  v8f acc[NT] = {};
#pragma unroll
  for (int t = 0; t < KT; ++t) {
    v16bf a;
#pragma unroll
    for (int r = 0; r < 2; ++r) {               // two contiguous 8-float runs (ISA A layout)
      const int base = t * 32 + r * 16;
      if (base + 16 <= K) {                     // compile-time foldable
        const float4 f0 = *(const float4*)(ap + base + hk);
        const float4 f1 = *(const float4*)(ap + base + hk + 4);
        a[r*8+0] = (__bf16)f0.x; a[r*8+1] = (__bf16)f0.y;
        a[r*8+2] = (__bf16)f0.z; a[r*8+3] = (__bf16)f0.w;
        a[r*8+4] = (__bf16)f1.x; a[r*8+5] = (__bf16)f1.y;
        a[r*8+6] = (__bf16)f1.z; a[r*8+7] = (__bf16)f1.w;
      } else {
#pragma unroll
        for (int j = 0; j < 8; ++j) a[r*8+j] = (__bf16)0.0f;
      }
    }
#pragma unroll
    for (int nt = 0; nt < NT; ++nt) {           // NT back-to-back WMMAs sharing A
      const int tb = (t * NT + nt) * 512;       // elements; half-region = 256
      const v8bf blo = *(const v8bf*)(bp + tb);
      const v8bf bhi = *(const v8bf*)(bp + tb + 256);
      const v16bf b = __builtin_shufflevector(blo, bhi,
                        0,1,2,3,4,5,6,7,8,9,10,11,12,13,14,15);
      acc[nt] = __builtin_amdgcn_wmma_f32_16x16x32_bf16(false, a, false, b,
                                                        (short)0, acc[nt], false, false);
    }
  }

#pragma unroll
  for (int nt = 0; nt < NT; ++nt) {
    const int col = nt * 16 + mrow;
    const float bb = bias[col];
#pragma unroll
    for (int rr = 0; rr < 8; ++rr) {
      const long long row = (long long)rb * 16 + rr + hk;   // C/D: VGPR rr <-> M=rr (+8 hi half)
      float v = acc[nt][rr] + bb;
      if (ACT) v = fmaxf(v, 0.0f);
      if (OUTBF16) ((bf16_t*)Cout)[row * ldc + col] = (__bf16)v;
      else         ((float*) Cout)[row * ldc + col] = v;
    }
  }
}

// ---------------- elementwise / scatter kernels ----------------
__global__ void k_zero_f32(float* p, long long n) {
  long long i = (long long)blockIdx.x * blockDim.x + threadIdx.x;
  if (i < n) p[i] = 0.0f;
}
__global__ void k_zero_i32(int* p, int n) {
  int i = blockIdx.x * blockDim.x + threadIdx.x;
  if (i < n) p[i] = 0;
}
__global__ void k_copy4(const float4* __restrict__ a, float4* __restrict__ b, long long n4) {
  long long i = (long long)blockIdx.x * blockDim.x + threadIdx.x;
  if (i < n4) b[i] = a[i];
}
__global__ void k_count(const int* __restrict__ batch, int* __restrict__ cnt) {
  int i = blockIdx.x * blockDim.x + threadIdx.x;
  if (i < N_NODES) atomicAdd(&cnt[batch[i]], 1);
}

// msg = relu(h[src] + ea);  z[dst] += msg.  One thread = one edge x 4 features.
__global__ void k_message(const float* __restrict__ h, const bf16_t* __restrict__ ea,
                          const int* __restrict__ src, const int* __restrict__ dst,
                          float* __restrict__ z)
{
  const long long total = (long long)N_EDGES * 16;
  const long long i = (long long)blockIdx.x * blockDim.x + threadIdx.x;
  if (i >= total) return;
  const int e = (int)(i >> 4);
  const int f = (int)(i & 15) * 4;
  const long long eb = (long long)e * HID + f;
  if (i + 16384 < total) __builtin_prefetch(&ea[eb + 65536], 0, 1); // global_prefetch_b8
  const float4 hv = *(const float4*)(h + (long long)src[e] * HID + f);
  const v4bf   ev = *(const v4bf*)(ea + eb);
  float* zp = z + (long long)dst[e] * HID + f;
  atomic_add_f32(zp + 0, fmaxf(hv.x + (float)ev[0], 0.0f));
  atomic_add_f32(zp + 1, fmaxf(hv.y + (float)ev[1], 0.0f));
  atomic_add_f32(zp + 2, fmaxf(hv.z + (float)ev[2], 0.0f));
  atomic_add_f32(zp + 3, fmaxf(hv.w + (float)ev[3], 0.0f));
}

// per-graph feature sums (C multiple of 4); one thread = one row x 4 features
__global__ void k_seg_sum4(const float* __restrict__ v, const int* __restrict__ batch,
                           float* __restrict__ gsum, int n, int C)
{
  const int c4 = C >> 2;
  const long long i = (long long)blockIdx.x * blockDim.x + threadIdx.x;
  if (i >= (long long)n * c4) return;
  const int row = (int)(i / c4);
  const int f   = (int)(i % c4) * 4;
  const float4 val = *(const float4*)(v + (long long)row * C + f);
  float* gp = gsum + (long long)batch[row] * C + f;
  atomic_add_f32(gp + 0, val.x);
  atomic_add_f32(gp + 1, val.y);
  atomic_add_f32(gp + 2, val.z);
  atomic_add_f32(gp + 3, val.w);
}

__global__ void k_mean(const float* __restrict__ gsum, const int* __restrict__ cnt,
                       float* __restrict__ gmean, int C)
{
  const int i = blockIdx.x * blockDim.x + threadIdx.x;
  if (i >= NGRAPH * C) return;
  gmean[i] = gsum[i] / fmaxf((float)cnt[i / C], 1.0f);
}

// GraphNorm pass 2: out = h2 - mean[batch]*gms ; accumulate out^2 per graph
__global__ void k_gn_center(const float* __restrict__ h2, const float* __restrict__ gmean,
                            const float* __restrict__ gms, const int* __restrict__ batch,
                            float* __restrict__ outc, float* __restrict__ gsum2)
{
  const long long i = (long long)blockIdx.x * blockDim.x + threadIdx.x;
  if (i >= (long long)N_NODES * HID) return;
  const int row = (int)(i >> 6);
  const int f   = (int)(i & 63);
  const int g   = batch[row];
  const float o = h2[i] - gmean[g * HID + f] * gms[f];
  outc[i] = o;
  atomic_add_f32(&gsum2[g * HID + f], o * o);
}

// GraphNorm pass 3: h = relu(gw*out/sqrt(var+eps)+gb) + h_in ; also write JK slot
__global__ void k_gn_final(const float* __restrict__ outc, const float* __restrict__ gvar,
                           const float* __restrict__ gw, const float* __restrict__ gb,
                           const int* __restrict__ batch, float* __restrict__ h,
                           float* __restrict__ hjk, int slot)
{
  const long long i = (long long)blockIdx.x * blockDim.x + threadIdx.x;
  if (i >= (long long)N_NODES * HID) return;
  const int row = (int)(i >> 6);
  const int f   = (int)(i & 63);
  const int g   = batch[row];
  float y = gw[f] * outc[i] * rsqrtf(gvar[g * HID + f] + GN_EPS) + gb[f];
  y = fmaxf(y, 0.0f) + h[i];
  h[i] = y;
  hjk[(long long)row * JKDIM + slot * HID + f] = y;
}

__global__ void k_copy_to_jk(const float* __restrict__ h, float* __restrict__ hjk, int slot) {
  const long long i = (long long)blockIdx.x * blockDim.x + threadIdx.x;
  if (i >= (long long)N_NODES * HID) return;
  const int row = (int)(i >> 6);
  const int f   = (int)(i & 63);
  hjk[(long long)row * JKDIM + slot * HID + f] = h[i];
}

// h_total = concat(segment_mean(hjk), graph_feat)
__global__ void k_htotal(const float* __restrict__ gsum, const int* __restrict__ cnt,
                         const float* __restrict__ gfeat, float* __restrict__ ht)
{
  const int i = blockIdx.x * blockDim.x + threadIdx.x;
  if (i >= NGRAPH * TOTDIM) return;
  const int g = i / TOTDIM;
  const int f = i % TOTDIM;
  if (f < JKDIM) ht[i] = gsum[(long long)g * JKDIM + f] / fmaxf((float)cnt[g], 1.0f);
  else           ht[i] = gfeat[g * GF_D + (f - JKDIM)];
}

// BatchNorm over rows (per column) + relu.  One block per column.
__global__ void k_bn_relu(const float* __restrict__ x, const float* __restrict__ gamma,
                          const float* __restrict__ beta, float* __restrict__ y,
                          int rows, int cols)
{
  __shared__ float ss[256], sq[256];
  const int c = blockIdx.x;
  float s = 0.0f, q = 0.0f;
  for (int r = threadIdx.x; r < rows; r += blockDim.x) {
    const float v = x[(long long)r * cols + c];
    s += v; q += v * v;
  }
  ss[threadIdx.x] = s; sq[threadIdx.x] = q;
  __syncthreads();
  for (int o = blockDim.x >> 1; o > 0; o >>= 1) {
    if ((int)threadIdx.x < o) { ss[threadIdx.x] += ss[threadIdx.x + o];
                                sq[threadIdx.x] += sq[threadIdx.x + o]; }
    __syncthreads();
  }
  const float mu  = ss[0] / rows;
  const float var = sq[0] / rows - mu * mu;
  const float inv = rsqrtf(var + BN_EPS);
  const float g = gamma[c], b = beta[c];
  for (int r = threadIdx.x; r < rows; r += blockDim.x) {
    const float v = x[(long long)r * cols + c];
    y[(long long)r * cols + c] = fmaxf(g * (v - mu) * inv + b, 0.0f);
  }
}

// Final [G,32]@[32,1]+b : one wave (32 lanes) per graph, lane-parallel dot.
__global__ void k_final(const float* __restrict__ t3, const float* __restrict__ w4,
                        const float* __restrict__ b4, float* __restrict__ out)
{
  const int wave = threadIdx.x >> 5;
  const int lane = threadIdx.x & 31;
  const int g = blockIdx.x * (blockDim.x >> 5) + wave;
  float v = (g < NGRAPH) ? t3[g * 32 + lane] * w4[lane] : 0.0f;
#pragma unroll
  for (int o = 16; o > 0; o >>= 1) v += __shfl_xor(v, o, 32);
  if (lane == 0 && g < NGRAPH) out[g] = v + b4[0];
}

// ---------------------------------------------------------------------------
extern "C" void kernel_launch(void* const* d_in, const int* in_sizes, int n_in,
                              void* d_out, int out_size, void* d_ws, size_t ws_size,
                              hipStream_t stream)
{
  // --- input pointers (top-level insertion order; params pytree sorted-key order) ---
  const float* x         = (const float*)d_in[0];
  const float* edge_attr = (const float*)d_in[1];
  const float* gfeat     = (const float*)d_in[2];
  const float *cw1[4], *cb1[4], *cw2[4], *cb2[4];
  for (int l = 0; l < 4; ++l) {
    cw1[l] = (const float*)d_in[3 + l*4 + 0];
    cb1[l] = (const float*)d_in[3 + l*4 + 1];
    cw2[l] = (const float*)d_in[3 + l*4 + 2];
    cb2[l] = (const float*)d_in[3 + l*4 + 3];
  }
  const float* eeW  = (const float*)d_in[19];
  const float* eeB  = (const float*)d_in[20];
  const float* bn1g = (const float*)d_in[21]; const float* bn1b = (const float*)d_in[22];
  const float* bn2g = (const float*)d_in[23]; const float* bn2b = (const float*)d_in[24];
  const float* l1W  = (const float*)d_in[25]; const float* l1B  = (const float*)d_in[26];
  const float* l2W  = (const float*)d_in[27]; const float* l2B  = (const float*)d_in[28];
  const float* l3W  = (const float*)d_in[29]; const float* l3B  = (const float*)d_in[30];
  const float* l4W  = (const float*)d_in[31]; const float* l4B  = (const float*)d_in[32];
  const float* neW  = (const float*)d_in[33]; const float* neB  = (const float*)d_in[34];
  const float *gw[4], *gb[4], *gms[4];
  for (int l = 0; l < 4; ++l) {
    gw[l]  = (const float*)d_in[35 + l*3 + 0];
    gb[l]  = (const float*)d_in[35 + l*3 + 1];
    gms[l] = (const float*)d_in[35 + l*3 + 2];
  }
  const int* eidx  = (const int*)d_in[47];
  const int* srcI  = eidx;
  const int* dstI  = eidx + N_EDGES;
  const int* batch = (const int*)d_in[48];

  // --- workspace carve (256B aligned) ---
  char* p = (char*)d_ws;
  auto alloc = [&](size_t bytes) -> char* {
    char* r = p; p += (bytes + 255) & ~(size_t)255; return r;
  };
  bf16_t* ea    = (bf16_t*)alloc((size_t)N_EDGES * HID * sizeof(bf16_t));   // 205 MB bf16 edge stream
  float*  h     = (float*) alloc((size_t)N_NODES * HID * 4);
  float*  z     = (float*) alloc((size_t)N_NODES * HID * 4);
  float*  t1    = (float*) alloc((size_t)N_NODES * HID * 4);
  float*  h2    = (float*) alloc((size_t)N_NODES * HID * 4);
  float*  hjk   = (float*) alloc((size_t)N_NODES * JKDIM * 4);
  float*  gsumJ = (float*) alloc((size_t)NGRAPH * JKDIM * 4);
  float*  gsum  = (float*) alloc((size_t)NGRAPH * HID * 4);
  float*  gmean = (float*) alloc((size_t)NGRAPH * HID * 4);
  float*  gvar  = (float*) alloc((size_t)NGRAPH * HID * 4);
  int*    cnt   = (int*)   alloc((size_t)NGRAPH * 4);
  float*  ht    = (float*) alloc((size_t)NGRAPH * TOTDIM * 4);
  float*  tg1   = (float*) alloc((size_t)NGRAPH * HID * 4);
  float*  tg1n  = (float*) alloc((size_t)NGRAPH * HID * 4);
  float*  tg2   = (float*) alloc((size_t)NGRAPH * HID * 4);
  float*  tg2n  = (float*) alloc((size_t)NGRAPH * HID * 4);
  float*  tg3   = (float*) alloc((size_t)NGRAPH * 32 * 4);

  auto nb = [](long long n, int t) { return (unsigned)((n + t - 1) / t); };
  const dim3 B128(128), B256(256);
  const unsigned GB_E = nb(N_EDGES / 16, 4);   // 25000 blocks (4 row-blocks each)
  const unsigned GB_N = nb(N_NODES / 16, 4);   // 1563
  const unsigned GB_G = nb(NGRAPH  / 16, 4);   // 16
  const long long NH  = (long long)N_NODES * HID;

  // per-graph node counts
  k_zero_i32<<<nb(NGRAPH,256), B256, 0, stream>>>(cnt, NGRAPH);
  k_count   <<<nb(N_NODES,256), B256, 0, stream>>>(batch, cnt);

  // encoders: ea = edge_attr@We+be (bf16 out);  h = x@Wn+bn
  k_gemm<EDGE_D,4,0,1><<<GB_E, B128, 0, stream>>>(edge_attr, EDGE_D, eeW, HID, eeB,
                                                  ea, HID, N_EDGES);
  k_gemm<NODE_D,4,0,0><<<GB_N, B128, 0, stream>>>(x, NODE_D, neW, HID, neB,
                                                  h, HID, N_NODES);
  k_copy_to_jk<<<nb(NH,256), B256, 0, stream>>>(h, hjk, 0);

  for (int l = 0; l < LAYERS; ++l) {
    // z = h ; z[dst] += relu(h[src] + ea)
    k_copy4  <<<nb(NH/4,256), B256, 0, stream>>>((const float4*)h, (float4*)z, NH/4);
    k_message<<<nb((long long)N_EDGES*16,256), B256, 0, stream>>>(h, ea, srcI, dstI, z);
    // t1 = relu(z@w1+b1) ; h2 = t1@w2+b2
    k_gemm<HID,4,1,0><<<GB_N, B128, 0, stream>>>(z,  HID, cw1[l], HID, cb1[l], t1, HID, N_NODES);
    k_gemm<HID,4,0,0><<<GB_N, B128, 0, stream>>>(t1, HID, cw2[l], HID, cb2[l], h2, HID, N_NODES);
    // GraphNorm
    k_zero_f32 <<<nb(NGRAPH*HID,256), B256, 0, stream>>>(gsum, NGRAPH*HID);
    k_seg_sum4 <<<nb(NH/4,256), B256, 0, stream>>>(h2, batch, gsum, N_NODES, HID);
    k_mean     <<<nb(NGRAPH*HID,256), B256, 0, stream>>>(gsum, cnt, gmean, HID);
    k_zero_f32 <<<nb(NGRAPH*HID,256), B256, 0, stream>>>(gsum, NGRAPH*HID);
    k_gn_center<<<nb(NH,256), B256, 0, stream>>>(h2, gmean, gms[l], batch, t1, gsum);
    k_mean     <<<nb(NGRAPH*HID,256), B256, 0, stream>>>(gsum, cnt, gvar, HID);
    k_gn_final <<<nb(NH,256), B256, 0, stream>>>(t1, gvar, gw[l], gb[l], batch, h, hjk, l + 1);
  }

  // graph readout + MLP head
  k_zero_f32<<<nb((long long)NGRAPH*JKDIM,256), B256, 0, stream>>>(gsumJ, (long long)NGRAPH*JKDIM);
  k_seg_sum4<<<nb((long long)N_NODES*(JKDIM/4),256), B256, 0, stream>>>(hjk, batch, gsumJ, N_NODES, JKDIM);
  k_htotal  <<<nb(NGRAPH*TOTDIM,256), B256, 0, stream>>>(gsumJ, cnt, gfeat, ht);

  k_gemm<TOTDIM,4,0,0><<<GB_G, B128, 0, stream>>>(ht,   TOTDIM, l1W, HID, l1B, tg1, HID, NGRAPH);
  k_bn_relu<<<HID, B256, 0, stream>>>(tg1, bn1g, bn1b, tg1n, NGRAPH, HID);
  k_gemm<HID,4,0,0>   <<<GB_G, B128, 0, stream>>>(tg1n, HID, l2W, HID, l2B, tg2, HID, NGRAPH);
  k_bn_relu<<<HID, B256, 0, stream>>>(tg2, bn2g, bn2b, tg2n, NGRAPH, HID);
  k_gemm<HID,2,1,0>   <<<GB_G, B128, 0, stream>>>(tg2n, HID, l3W, 32, l3B, tg3, 32, NGRAPH);
  k_final<<<NGRAPH/8, B256, 0, stream>>>(tg3, l4W, l4B, (float*)d_out);
}
